// GATEAU_59012850647619
// MI455X (gfx1250) — matrix operand order repaired and checked by reading, not compiled, not run on hardware
//
#include <hip/hip_runtime.h>
#include <math.h>

#define DIM 128

typedef __attribute__((ext_vector_type(16))) __bf16        v16bf;
typedef __attribute__((ext_vector_type(8)))  float         v8f;
typedef __attribute__((ext_vector_type(4)))  unsigned int  v4u;
typedef int vint4 __attribute__((vector_size(16)));

#if __has_builtin(__builtin_amdgcn_global_load_async_to_lds_b128) && \
    __has_builtin(__builtin_amdgcn_s_wait_asynccnt)
#define HAS_ASYNC_LDS 1
#else
#define HAS_ASYNC_LDS 0
#endif

// round-to-nearest-even f32 -> bf16 (returned in low 16 bits)
__device__ __forceinline__ unsigned int bf16rne(float x) {
    unsigned int u = __float_as_uint(x);
    unsigned int r = u + 0x7FFFu + ((u >> 16) & 1u);
    return r >> 16;
}

union Frag16 { v4u q[2]; v16bf v; };

// Build a 16-element bf16 A/B fragment from two runs of 8 consecutive f32 in LDS
__device__ __forceinline__ v16bf frag_from_f32(const float* p0, const float* p1) {
    float4 x0 = ((const float4*)p0)[0], x1 = ((const float4*)p0)[1];
    float4 y0 = ((const float4*)p1)[0], y1 = ((const float4*)p1)[1];
    Frag16 f;
    f.q[0][0] = bf16rne(x0.x) | (bf16rne(x0.y) << 16);
    f.q[0][1] = bf16rne(x0.z) | (bf16rne(x0.w) << 16);
    f.q[0][2] = bf16rne(x1.x) | (bf16rne(x1.y) << 16);
    f.q[0][3] = bf16rne(x1.z) | (bf16rne(x1.w) << 16);
    f.q[1][0] = bf16rne(y0.x) | (bf16rne(y0.y) << 16);
    f.q[1][1] = bf16rne(y0.z) | (bf16rne(y0.w) << 16);
    f.q[1][2] = bf16rne(y1.x) | (bf16rne(y1.y) << 16);
    f.q[1][3] = bf16rne(y1.z) | (bf16rne(y1.w) << 16);
    return f.v;
}

// bf16 fragment straight from LDS (two 16B chunks)
__device__ __forceinline__ v16bf ld_frag(const unsigned short* p0,
                                         const unsigned short* p1) {
    Frag16 f;
    f.q[0] = *(const v4u*)p0;
    f.q[1] = *(const v4u*)p1;
    return f.v;
}

// async global->LDS 16B copy (falls back to synchronous copy)
__device__ __forceinline__ void cp_b128(const void* g, void* l) {
#if HAS_ASYNC_LDS
    __builtin_amdgcn_global_load_async_to_lds_b128(
        (__attribute__((address_space(1))) vint4*)g,
        (__attribute__((address_space(3))) vint4*)l, 0, 0);
#else
    *(v4u*)l = *(const v4u*)g;
#endif
}

__device__ __forceinline__ void cp_wait() {
#if HAS_ASYNC_LDS
    __builtin_amdgcn_s_wait_asynccnt(0);
#endif
}

__device__ __forceinline__ void atomicMaxFloat(float* addr, float v) {
    int vi = __float_as_int(v);
    if (vi >= 0) atomicMax((int*)addr, vi);
    else         atomicMin((unsigned int*)addr, (unsigned int)vi);
}

// ---------------------------------------------------------------------------
// K0: convert W1,W2,Wm,We (f32, [K=128][N=128]) -> bf16 transposed [N][K]
// ---------------------------------------------------------------------------
__global__ void kwconv(const float* __restrict__ W1, const float* __restrict__ W2,
                       const float* __restrict__ Wm, const float* __restrict__ We,
                       unsigned short* __restrict__ Wt) {
    int t = blockIdx.x * 256 + threadIdx.x;       // 16384 = 128*128
    if (t >= DIM * DIM) return;
    int n = t >> 7, k = t & 127;
    const float* srcs[4] = {W1, W2, Wm, We};
    for (int m = 0; m < 4; ++m)
        Wt[m * DIM * DIM + n * DIM + k] =
            (unsigned short)bf16rne(srcs[m][k * DIM + n]);
}

// ---------------------------------------------------------------------------
// K1: zero output node accumulator, init segment max/sum
// ---------------------------------------------------------------------------
__global__ void kinit(float* __restrict__ out_nodes, float* __restrict__ nmax,
                      float* __restrict__ nsum, int N) {
    long long i = (long long)blockIdx.x * 256 + threadIdx.x;
    if (i < (long long)N * DIM) out_nodes[i] = 0.0f;
    if (i < N) { nmax[i] = -INFINITY; nsum[i] = 0.0f; }
}

// ---------------------------------------------------------------------------
// K2: node projections  P1 = nodes@W1+b1, P2 = nodes@W2+b2, PM = nodes@Wm+bm
//     128-row tile / block, 8 waves, WMMA bf16 (f32 accumulate)
// ---------------------------------------------------------------------------
__global__ __launch_bounds__(256)
void knodeproj(const float* __restrict__ nodes, int N,
               const unsigned short* __restrict__ Wt,      // 3 matrices bf16 [N][K]
               const float* __restrict__ b1, const float* __restrict__ b2,
               const float* __restrict__ bm,
               float* __restrict__ P1, float* __restrict__ P2,
               float* __restrict__ PM) {
    __shared__ float          sAf[DIM * DIM];  // 64KB: 128 activation rows, f32
    __shared__ unsigned short sB[DIM * DIM];   // 32KB: one W^T, bf16

    const int  tid  = threadIdx.x;
    const int  row0 = blockIdx.x * 128;
    const bool full = (row0 + 128 <= N);

    // stage activation tile -> LDS via async DMA (raw f32)
    for (int i = tid; i < 128 * 32; i += 256) {          // 32 float4 per row
        int r  = i >> 5;
        int c4 = (i & 31) * 4;
        int gr = row0 + r; if (gr >= N) gr = N - 1;
        __builtin_prefetch(nodes + (size_t)(gr + 128) * DIM + c4, 0, 1);
        cp_b128(nodes + (size_t)gr * DIM + c4, &sAf[r * DIM + c4]);
    }
    cp_wait();
    __syncthreads();

    const int wave = tid >> 5;
    const int lane = tid & 31;
    const int M    = lane & 15;
    const int hi   = lane >> 4;
    const int arow = wave * 16 + M;

    // A fragments (bf16) for this wave's 16 rows: K = kt*32 + hi*8 (+16)
    v16bf aF[4];
    for (int kt = 0; kt < 4; ++kt) {
        int kb = kt * 32 + hi * 8;
        aF[kt] = frag_from_f32(&sAf[arow * DIM + kb], &sAf[arow * DIM + kb + 16]);
    }

    for (int m = 0; m < 3; ++m) {
        __syncthreads();
        const v4u* src = (const v4u*)(Wt + (size_t)m * DIM * DIM);
        for (int i = tid; i < 2048; i += 256)
            cp_b128(src + i, (v4u*)sB + i);
        cp_wait();
        __syncthreads();

        const float* bias = (m == 0) ? b1 : (m == 1) ? b2 : bm;
        float* out        = (m == 0) ? P1 : (m == 1) ? P2 : PM;

        for (int ct = 0; ct < 8; ++ct) {
            int ncol = ct * 16 + M;
            // preload all B fragments for this tile, then run the WMMA chain
            v16bf bF[4];
            for (int kt = 0; kt < 4; ++kt) {
                int kb = kt * 32 + hi * 8;
                bF[kt] = ld_frag(&sB[ncol * DIM + kb], &sB[ncol * DIM + kb + 16]);
            }
            v8f acc = {};
            for (int kt = 0; kt < 4; ++kt)
                acc = __builtin_amdgcn_wmma_f32_16x16x32_bf16(
                    false, aF[kt], false, bF[kt], (short)0, acc, false, false);

            int    col  = ct * 16 + M;
            float  bcl  = bias[col];
            size_t base = (size_t)(row0 + wave * 16 + hi * 8) * DIM + col;
            if (full) {
                for (int r = 0; r < 8; ++r)
                    out[base + (size_t)r * DIM] = acc[r] + bcl;
            } else {
                for (int r = 0; r < 8; ++r) {
                    int grow = row0 + wave * 16 + r + hi * 8;
                    if (grow < N) out[(size_t)grow * DIM + col] = acc[r] + bcl;
                }
            }
        }
    }
}

// ---------------------------------------------------------------------------
// K3: edge_feats = edges@We + be + P1[senders] + P2[receivers]   (WMMA)
// ---------------------------------------------------------------------------
__global__ __launch_bounds__(256)
void kedge(const float* __restrict__ edges, int E, int N,
           const unsigned short* __restrict__ Wet, const float* __restrict__ be,
           const int* __restrict__ senders, const int* __restrict__ receivers,
           const float* __restrict__ P1, const float* __restrict__ P2,
           float* __restrict__ edge_feats) {
    __shared__ float          sAf[DIM * DIM];  // 64KB: 128 edge rows, f32
    __shared__ unsigned short sB[DIM * DIM];   // 32KB: We^T, bf16

    const int  tid  = threadIdx.x;
    const int  e0   = blockIdx.x * 128;
    const bool full = (e0 + 128 <= E);

    for (int i = tid; i < 128 * 32; i += 256) {
        int r  = i >> 5;
        int c4 = (i & 31) * 4;
        int ge = e0 + r; if (ge >= E) ge = E - 1;
        __builtin_prefetch(edges + (size_t)(ge + 128) * DIM + c4, 0, 1);
        cp_b128(edges + (size_t)ge * DIM + c4, &sAf[r * DIM + c4]);
    }
    for (int i = tid; i < 2048; i += 256)
        cp_b128((const v4u*)Wet + i, (v4u*)sB + i);
    cp_wait();
    __syncthreads();

    const int wave = tid >> 5;
    const int lane = tid & 31;
    const int M    = lane & 15;
    const int hi   = lane >> 4;
    const int arow = wave * 16 + M;

    v16bf aF[4];
    for (int kt = 0; kt < 4; ++kt) {
        int kb = kt * 32 + hi * 8;
        aF[kt] = frag_from_f32(&sAf[arow * DIM + kb], &sAf[arow * DIM + kb + 16]);
    }

    // per-lane edge indices for the epilogue (rows r -> M = r + hi*8)
    int eid[8], sI[8], rI[8];
    for (int r = 0; r < 8; ++r) {
        int e  = e0 + wave * 16 + r + hi * 8;
        int ec = (e < E) ? e : (E - 1);
        eid[r] = ec;
        sI[r]  = senders[ec];
        rI[r]  = receivers[ec];
    }

    for (int ct = 0; ct < 8; ++ct) {
        int ncol = ct * 16 + M;
        v16bf bF[4];
        for (int kt = 0; kt < 4; ++kt) {
            int kb = kt * 32 + hi * 8;
            bF[kt] = ld_frag(&sB[ncol * DIM + kb], &sB[ncol * DIM + kb + 16]);
        }
        v8f acc = {};
        for (int kt = 0; kt < 4; ++kt)
            acc = __builtin_amdgcn_wmma_f32_16x16x32_bf16(
                false, aF[kt], false, bF[kt], (short)0, acc, false, false);

        int   col = ct * 16 + M;
        float bcl = be[col];
        if (full) {
            for (int r = 0; r < 8; ++r) {
                float v = acc[r] + bcl + P1[(size_t)sI[r] * DIM + col]
                                  + P2[(size_t)rI[r] * DIM + col];
                edge_feats[(size_t)eid[r] * DIM + col] = v;
            }
        } else {
            for (int r = 0; r < 8; ++r) {
                int e = e0 + wave * 16 + r + hi * 8;
                if (e >= E) continue;
                float v = acc[r] + bcl + P1[(size_t)sI[r] * DIM + col]
                                  + P2[(size_t)rI[r] * DIM + col];
                edge_feats[(size_t)e * DIM + col] = v;
            }
        }
    }
}

// ---------------------------------------------------------------------------
// K4: attention logits (leaky_relu(ef@Wa + ba)) + segment max (one wave/edge)
// ---------------------------------------------------------------------------
__global__ __launch_bounds__(256)
void katt(const float* __restrict__ edge_feats, const float* __restrict__ Wa,
          const float* __restrict__ ba, const int* __restrict__ receivers,
          int E, float* __restrict__ att, float* __restrict__ nmax) {
    int wave = threadIdx.x >> 5, lane = threadIdx.x & 31;
    long long e = (long long)blockIdx.x * 8 + wave;
    if (e >= E) return;
    float4 f = *(const float4*)(edge_feats + e * DIM + lane * 4);
    float4 w = *(const float4*)(Wa + lane * 4);
    float p = f.x * w.x + f.y * w.y + f.z * w.z + f.w * w.w;
    for (int off = 16; off; off >>= 1) p += __shfl_xor(p, off, 32);
    if (lane == 0) {
        float logit = p + ba[0];
        logit = logit > 0.0f ? logit : 0.01f * logit;   // leaky_relu slope 0.01
        att[e] = logit;
        atomicMaxFloat(&nmax[receivers[e]], logit);
    }
}

// ---------------------------------------------------------------------------
// K5: e = exp(logit - max[recv]); segment sum
// ---------------------------------------------------------------------------
__global__ void kexp(const int* __restrict__ receivers, float* __restrict__ att,
                     const float* __restrict__ nmax, float* __restrict__ nsum,
                     int E) {
    long long e = (long long)blockIdx.x * 256 + threadIdx.x;
    if (e >= E) return;
    int rc  = receivers[e];
    float v = expf(att[e] - nmax[rc]);
    att[e]  = v;
    atomicAdd(&nsum[rc], v);
}

// ---------------------------------------------------------------------------
// K6: new_nodes[recv] += (att/sum[recv]) * PM[sender]   (one wave/edge)
// ---------------------------------------------------------------------------
__global__ __launch_bounds__(256)
void kscat(const int* __restrict__ senders, const int* __restrict__ receivers,
           const float* __restrict__ att, const float* __restrict__ nsum,
           const float* __restrict__ PM, float* __restrict__ out_nodes, int E) {
    int wave = threadIdx.x >> 5, lane = threadIdx.x & 31;
    long long e = (long long)blockIdx.x * 8 + wave;
    if (e >= E) return;
    int s = senders[e], rc = receivers[e];
    float w  = att[e] / nsum[rc];
    float4 m = *(const float4*)(PM + (size_t)s * DIM + lane * 4);
    float* dst = out_nodes + (size_t)rc * DIM + lane * 4;
    atomicAdd(dst + 0, w * m.x);
    atomicAdd(dst + 1, w * m.y);
    atomicAdd(dst + 2, w * m.z);
    atomicAdd(dst + 3, w * m.w);
}

// ---------------------------------------------------------------------------
extern "C" void kernel_launch(void* const* d_in, const int* in_sizes, int n_in,
                              void* d_out, int out_size, void* d_ws, size_t ws_size,
                              hipStream_t stream) {
    const float* nodes     = (const float*)d_in[0];
    const float* edges     = (const float*)d_in[1];
    const int*   senders   = (const int*)d_in[2];
    const int*   receivers = (const int*)d_in[3];
    const float* W1 = (const float*)d_in[4];
    const float* b1 = (const float*)d_in[5];
    const float* W2 = (const float*)d_in[6];
    const float* b2 = (const float*)d_in[7];
    const float* We = (const float*)d_in[8];
    const float* be = (const float*)d_in[9];
    const float* Wa = (const float*)d_in[10];
    const float* ba = (const float*)d_in[11];
    const float* Wm = (const float*)d_in[12];
    const float* bm = (const float*)d_in[13];

    const int N = in_sizes[0] / DIM;
    const int E = in_sizes[1] / DIM;

    float* out_nodes = (float*)d_out;                       // [N,128]
    float* out_ef    = out_nodes + (size_t)N * DIM;         // [E,128]

    float* P1   = (float*)d_ws;                             // [N,128]
    float* P2   = P1 + (size_t)N * DIM;                     // [N,128]
    float* PM   = P2 + (size_t)N * DIM;                     // [N,128]
    float* att  = PM + (size_t)N * DIM;                     // [E]
    float* nmax = att + E;                                  // [N]
    float* nsum = nmax + N;                                 // [N]
    unsigned short* Wt = (unsigned short*)(nsum + N);       // 4 x 128*128 bf16

    kwconv<<<(DIM * DIM + 255) / 256, 256, 0, stream>>>(W1, W2, Wm, We, Wt);
    kinit<<<(int)(((long long)N * DIM + 255) / 256), 256, 0, stream>>>(
        out_nodes, nmax, nsum, N);
    knodeproj<<<(N + 127) / 128, 256, 0, stream>>>(
        nodes, N, Wt, b1, b2, bm, P1, P2, PM);
    kedge<<<(E + 127) / 128, 256, 0, stream>>>(
        edges, E, N, Wt + 3 * DIM * DIM, be, senders, receivers, P1, P2, out_ef);
    katt<<<(E + 7) / 8, 256, 0, stream>>>(out_ef, Wa, ba, receivers, E, att, nmax);
    kexp<<<(E + 255) / 256, 256, 0, stream>>>(receivers, att, nmax, nsum, E);
    kscat<<<(E + 7) / 8, 256, 0, stream>>>(senders, receivers, att, nsum, PM,
                                           out_nodes, E);
}
